// AttentionBlock_8684423873092
// MI455X (gfx1250) — matrix operand order repaired and checked by reading, not compiled, not run on hardware
//
#include <hip/hip_runtime.h>

// ---------------------------------------------------------------------------
// AttentionBlock for MI455X (gfx1250, wave32, WMMA 16x16x32 f16->f32).
// Needs ~24.2 MB of workspace (d_ws).
// ---------------------------------------------------------------------------

typedef __attribute__((ext_vector_type(16))) _Float16 v16h;
typedef __attribute__((ext_vector_type(8)))  _Float16 v8h;
typedef __attribute__((ext_vector_type(8)))  float    v8f;

#define SEQ   9216
#define CCH   256
#define NH    4
#define HD    64
#define NG    32
#define GELEM (8 * SEQ)           // elements per group (8 ch * 9216)
#define LOG2E 1.4426950408889634f

static __device__ __forceinline__ v8f vzero8() {
  v8f z = {0.f, 0.f, 0.f, 0.f, 0.f, 0.f, 0.f, 0.f};
  return z;
}

static __device__ __forceinline__ v8f wmma16(v16h a, v16h b, v8f c) {
  return __builtin_amdgcn_wmma_f32_16x16x32_f16(false, a, false, b,
                                                (short)0, c, false, false);
}

// B-operand: 16 contiguous halfs (lane = column n, K = 16*half + e)
static __device__ __forceinline__ v16h ldB(const _Float16* p) {
  v8h lo = *(const v8h*)(p);
  v8h hi = *(const v8h*)(p + 8);
  return __builtin_shufflevector(lo, hi, 0,1,2,3,4,5,6,7,8,9,10,11,12,13,14,15);
}

// A-operand: 16 contiguous halfs (lane = row m, K = 16*half + e)
static __device__ __forceinline__ v16h ldA(const _Float16* p) {
  v8h lo = *(const v8h*)(p);
  v8h hi = *(const v8h*)(p + 8);
  return __builtin_shufflevector(lo, hi, 0,1,2,3,4,5,6,7,8,9,10,11,12,13,14,15);
}

// ---------------------------------------------------------------------------
// GroupNorm: per-group mean / rstd (32 groups, 73728 elems each, contiguous)
// ---------------------------------------------------------------------------
__global__ void gn_stats(const float* __restrict__ x, float* __restrict__ stats) {
  __shared__ float ssum[256];
  __shared__ float ssq[256];
  const int g = blockIdx.x, t = threadIdx.x;
  const float* p = x + (size_t)g * GELEM;
  float s = 0.f, q = 0.f;
  for (int i = t; i < GELEM; i += 256) { float v = p[i]; s += v; q += v * v; }
  ssum[t] = s; ssq[t] = q;
  __syncthreads();
  for (int off = 128; off > 0; off >>= 1) {
    if (t < off) { ssum[t] += ssum[t + off]; ssq[t] += ssq[t + off]; }
    __syncthreads();
  }
  if (t == 0) {
    float mean = ssum[0] / (float)GELEM;
    float var  = ssq[0] / (float)GELEM - mean * mean;
    stats[g * 2]     = mean;
    stats[g * 2 + 1] = rsqrtf(var + 1e-5f);
  }
}

// normalize + affine, transpose [c][s] -> token-major f16 [s][c]
__global__ void gn_apply(const float* __restrict__ x, const float* __restrict__ stats,
                         const float* __restrict__ nw, const float* __restrict__ nb,
                         _Float16* __restrict__ xnf) {
  const int tid = blockIdx.x * blockDim.x + threadIdx.x;  // == c*SEQ + s
  const int c = tid / SEQ, s = tid % SEQ;
  const int g = c >> 3;
  float v = (x[tid] - stats[g * 2]) * stats[g * 2 + 1] * nw[c] + nb[c];
  xnf[(size_t)s * CCH + c] = (_Float16)v;
}

__global__ void f32_to_f16(const float* __restrict__ src, _Float16* __restrict__ dst, int n) {
  int i = blockIdx.x * blockDim.x + threadIdx.x;
  if (i < n) dst[i] = (_Float16)src[i];
}

// ---------------------------------------------------------------------------
// QKV GEMM: [9216 x 256] x [256 x 768].  One wave = 16 tokens x 64 features.
// Q,K stored [h][s][64]; V stored transposed [h][64][s].
// ---------------------------------------------------------------------------
__global__ void __launch_bounds__(128) qkv_gemm(
    const _Float16* __restrict__ xnf, const _Float16* __restrict__ wq,
    const float* __restrict__ bias, _Float16* __restrict__ qb,
    _Float16* __restrict__ kb, _Float16* __restrict__ vt) {
  const int lane = threadIdx.x & 31;
  const int wave = (blockIdx.x * blockDim.x + threadIdx.x) >> 5;
  const int mt = wave / 12, nbk = wave % 12;
  const int t0 = mt * 16, n0 = nbk * 64;
  const int mn = lane & 15, hlf = lane >> 4;

  v8f acc[4];
#pragma unroll
  for (int i = 0; i < 4; ++i) acc[i] = vzero8();

  const _Float16* arow = xnf + (size_t)(t0 + mn) * CCH + hlf * 16;
  const _Float16* brow = wq + (size_t)(n0 + mn) * CCH + hlf * 16;

  for (int ck = 0; ck < CCH; ck += 32) {
    v16h a = ldA(arow + ck);
#pragma unroll
    for (int nt = 0; nt < 4; ++nt)
      acc[nt] = wmma16(a, ldB(brow + (size_t)nt * 16 * CCH + ck), acc[nt]);
  }

  const int row0 = t0 + hlf * 8;
#pragma unroll
  for (int nt = 0; nt < 4; ++nt) {
    const int f = n0 + nt * 16 + mn;
    const float bval = bias[f];
#pragma unroll
    for (int r = 0; r < 8; ++r) {
      const int s = row0 + r;
      const _Float16 hv = (_Float16)(acc[nt][r] + bval);
      if (f < 256) {                    // Q  (uniform branch per wave)
        qb[((size_t)((f >> 6) * SEQ + s)) * HD + (f & 63)] = hv;
      } else if (f < 512) {             // K
        const int fk = f - 256;
        kb[((size_t)((fk >> 6) * SEQ + s)) * HD + (fk & 63)] = hv;
      } else {                          // V (transposed)
        const int fv = f - 512;
        vt[((size_t)((fv >> 6) * HD + (fv & 63))) * SEQ + s] = hv;
      }
    }
  }
}

// ---------------------------------------------------------------------------
// Tiled flash attention: one wave = one head x 16 queries, 32 keys / step.
// ---------------------------------------------------------------------------
__global__ void __launch_bounds__(128) attn_kernel(
    const _Float16* __restrict__ qb, const _Float16* __restrict__ kb,
    const _Float16* __restrict__ vt, _Float16* __restrict__ ao) {
  __shared__ __align__(32) _Float16 plds[4][16 * 32];   // wave-private P tiles
  const int lane = threadIdx.x & 31;
  const int wslot = threadIdx.x >> 5;
  const int wave = (blockIdx.x * blockDim.x + threadIdx.x) >> 5;
  const int head = wave / 576;
  const int q0 = (wave % 576) * 16;
  const int tile = q0 >> 12;                               // /4096
  const int k_lo = (tile == 0) ? 0 : (tile == 1 ? 2048 : 6144);
  const int k_hi = (tile == 0) ? 6144 : 9216;
  const int mn = lane & 15, hlf = lane >> 4;

  // Q in A-layout, softmax scale (1/8) folded in once.
  const _Float16* qrow = qb + ((size_t)(head * SEQ + q0 + mn)) * HD + hlf * 16;
  v16h aq0 = ldA(qrow);
  v16h aq1 = ldA(qrow + 32);
#pragma unroll
  for (int e = 0; e < 16; ++e) {
    aq0[e] *= (_Float16)0.125f;
    aq1[e] *= (_Float16)0.125f;
  }

  float mrun[8], lrun[8];
#pragma unroll
  for (int r = 0; r < 8; ++r) { mrun[r] = -3.0e38f; lrun[r] = 0.f; }
  v8f o0 = vzero8(), o1 = vzero8(), o2 = vzero8(), o3 = vzero8();

  const _Float16* kbase = kb + (size_t)head * SEQ * HD;
  const _Float16* vbase = vt + (size_t)head * HD * SEQ;
  _Float16* pl = &plds[wslot][0];

  for (int kk = k_lo; kk < k_hi; kk += 32) {
    // ---- S = (Q*scale) K^T : two 16x16 tiles over 32 keys, K-dim = d = 64
    const _Float16* kr0 = kbase + (size_t)(kk + mn) * HD + hlf * 16;
    const _Float16* kr1 = kbase + (size_t)(kk + 16 + mn) * HD + hlf * 16;
    v8f s0 = vzero8(), s1 = vzero8();
    s0 = wmma16(aq0, ldB(kr0), s0);
    s0 = wmma16(aq1, ldB(kr0 + 32), s0);
    s1 = wmma16(aq0, ldB(kr1), s1);
    s1 = wmma16(aq1, ldB(kr1 + 32), s1);
    if (kk + 32 < k_hi)
      __builtin_prefetch(kbase + (size_t)(kk + 32 + mn) * HD, 0, 0);

    // ---- online softmax (row stats live per VGPR slot; reduce across 16 lanes)
    float corr[8];
#pragma unroll
    for (int r = 0; r < 8; ++r) {
      float mx = fmaxf(s0[r], s1[r]);
      mx = fmaxf(mx, __shfl_xor(mx, 1));
      mx = fmaxf(mx, __shfl_xor(mx, 2));
      mx = fmaxf(mx, __shfl_xor(mx, 4));
      mx = fmaxf(mx, __shfl_xor(mx, 8));
      const float mnew = fmaxf(mrun[r], mx);
      corr[r] = exp2f((mrun[r] - mnew) * LOG2E);
      const float p0 = exp2f((s0[r] - mnew) * LOG2E);
      const float p1 = exp2f((s1[r] - mnew) * LOG2E);
      float sm = p0 + p1;
      sm += __shfl_xor(sm, 1);
      sm += __shfl_xor(sm, 2);
      sm += __shfl_xor(sm, 4);
      sm += __shfl_xor(sm, 8);
      lrun[r] = lrun[r] * corr[r] + sm;
      mrun[r] = mnew;
      const int row = r + hlf * 8;                 // C/D -> row-major transpose
      pl[row * 32 + mn]      = (_Float16)p0;
      pl[row * 32 + 16 + mn] = (_Float16)p1;
      o0[r] *= corr[r]; o1[r] *= corr[r]; o2[r] *= corr[r]; o3[r] *= corr[r];
    }
    asm volatile("s_wait_dscnt 0" ::: "memory");   // P stores visible before A-load

    // ---- O += P V : A = P [16x32], B = V^T columns (contiguous along s)
    v16h ap = ldA(pl + mn * 32 + hlf * 16);
    asm volatile("" ::: "memory");
    const _Float16* vc = vbase + (size_t)mn * SEQ + kk + hlf * 16;
    o0 = wmma16(ap, ldB(vc), o0);
    o1 = wmma16(ap, ldB(vc + 16 * SEQ), o1);
    o2 = wmma16(ap, ldB(vc + 32 * SEQ), o2);
    o3 = wmma16(ap, ldB(vc + 48 * SEQ), o3);
  }

  // ---- normalize and store token-major f16 [s][256]
  const int row0 = q0 + hlf * 8;
#pragma unroll
  for (int r = 0; r < 8; ++r) {
    const float inv = 1.0f / lrun[r];
    const size_t base = (size_t)(row0 + r) * CCH + head * HD + mn;
    ao[base]      = (_Float16)(o0[r] * inv);
    ao[base + 16] = (_Float16)(o1[r] * inv);
    ao[base + 32] = (_Float16)(o2[r] * inv);
    ao[base + 48] = (_Float16)(o3[r] * inv);
  }
}

// ---------------------------------------------------------------------------
// Proj GEMM [9216 x 256] x [256 x 256] + bias + residual -> out [c][s] f32
// ---------------------------------------------------------------------------
__global__ void __launch_bounds__(128) proj_gemm(
    const _Float16* __restrict__ ao, const _Float16* __restrict__ wp,
    const float* __restrict__ pb, const float* __restrict__ x,
    float* __restrict__ out) {
  const int lane = threadIdx.x & 31;
  const int wave = (blockIdx.x * blockDim.x + threadIdx.x) >> 5;
  const int mt = wave >> 2, nbk = wave & 3;
  const int t0 = mt * 16, n0 = nbk * 64;
  const int mn = lane & 15, hlf = lane >> 4;

  v8f acc[4];
#pragma unroll
  for (int i = 0; i < 4; ++i) acc[i] = vzero8();

  const _Float16* arow = ao + (size_t)(t0 + mn) * CCH + hlf * 16;
  const _Float16* brow = wp + (size_t)(n0 + mn) * CCH + hlf * 16;
  for (int ck = 0; ck < CCH; ck += 32) {
    v16h a = ldA(arow + ck);
#pragma unroll
    for (int nt = 0; nt < 4; ++nt)
      acc[nt] = wmma16(a, ldB(brow + (size_t)nt * 16 * CCH + ck), acc[nt]);
  }

  const int row0 = t0 + hlf * 8;
#pragma unroll
  for (int nt = 0; nt < 4; ++nt) {
    const int c = n0 + nt * 16 + mn;
    const float bv = pb[c];
#pragma unroll
    for (int r = 0; r < 8; ++r) {
      const size_t idx = (size_t)c * SEQ + (row0 + r);
      out[idx] = acc[nt][r] + bv + x[idx];
    }
  }
}

// ---------------------------------------------------------------------------
extern "C" void kernel_launch(void* const* d_in, const int* in_sizes, int n_in,
                              void* d_out, int out_size, void* d_ws, size_t ws_size,
                              hipStream_t stream) {
  const float* x      = (const float*)d_in[0];
  const float* norm_w = (const float*)d_in[1];
  const float* norm_b = (const float*)d_in[2];
  const float* qkv_w  = (const float*)d_in[3];
  const float* qkv_b  = (const float*)d_in[4];
  const float* proj_w = (const float*)d_in[5];
  const float* proj_b = (const float*)d_in[6];
  float* out = (float*)d_out;

  char* ws = (char*)d_ws;
  _Float16* xnf  = (_Float16*)(ws);                 //  4718592 B: [s][256] f16
  _Float16* wq16 = (_Float16*)(ws + 4718592);       //   393216 B: qkv_w f16
  _Float16* wp16 = (_Float16*)(ws + 5111808);       //   131072 B: proj_w f16
  _Float16* qb   = (_Float16*)(ws + 5242880);       //  4718592 B: Q [h][s][64]
  _Float16* kb   = (_Float16*)(ws + 9961472);       //  4718592 B: K [h][s][64]
  _Float16* vt   = (_Float16*)(ws + 14680064);      //  4718592 B: V^T [h][64][s]
  _Float16* ao   = (_Float16*)(ws + 19398656);      //  4718592 B: attn out [s][256]
  float*    st   = (float*)(ws + 24117248);         //      256 B: gn stats

  gn_stats<<<NG, 256, 0, stream>>>(x, st);
  gn_apply<<<(SEQ * CCH) / 256, 256, 0, stream>>>(x, st, norm_w, norm_b, xnf);
  f32_to_f16<<<(768 * 256 + 255) / 256, 256, 0, stream>>>(qkv_w, wq16, 768 * 256);
  f32_to_f16<<<(256 * 256 + 255) / 256, 256, 0, stream>>>(proj_w, wp16, 256 * 256);
  qkv_gemm<<<1728, 128, 0, stream>>>(xnf, wq16, qkv_b, qb, kb, vt);   // 6912 waves
  attn_kernel<<<576, 128, 0, stream>>>(qb, kb, vt, ao);               // 2304 waves
  proj_gemm<<<576, 128, 0, stream>>>(ao, wp16, proj_b, x, out);       // 2304 waves
}
